// RNNExpertModel_61306363183413
// MI455X (gfx1250) — matrix operand order repaired and checked by reading, not compiled, not run on hardware
//
#include <hip/hip_runtime.h>
#include <stdint.h>

// ---------------------------------------------------------------------------
// 2-layer LSTM LM for gfx1250 (MI455X).  Memory-bound (~0.8 GB traffic,
// ~390 GFLOP): bf16 WMMA everywhere, fp32 accumulate, weights converted to
// bf16 once per call so fc_W (64 MB) lives in the 192 MB L2.
// Pipeline: async-to-LDS panel staging (ASYNCcnt DMA), software-pipelined
// k-loops with a peeled epilogue (no in-loop guards), templated K.
// ---------------------------------------------------------------------------

typedef __attribute__((ext_vector_type(16))) __bf16 v16bf;
typedef __attribute__((ext_vector_type(8)))  __bf16 v8bf;
typedef __attribute__((ext_vector_type(8)))  float  v8f;
typedef __attribute__((ext_vector_type(4)))  int    v4i;

#define BATCH 16
#define SEQ   256
#define EMB   512
#define HID   1024
#define NGATE 4096          // 4*HID
#define VOCAB 32000
#define NTOK  4096          // BATCH*SEQ

#if __has_builtin(__builtin_amdgcn_global_load_async_to_lds_b128) && \
    __has_builtin(__builtin_amdgcn_s_wait_asynccnt)
#define HAVE_ASYNC_LDS 1
#else
#define HAVE_ASYNC_LDS 0
#endif

static __device__ __forceinline__ __bf16 f2bf(float f) {
  unsigned u = __builtin_bit_cast(unsigned, f);
  u += 0x7FFFu + ((u >> 16) & 1u);          // round-to-nearest-even
  unsigned short h = (unsigned short)(u >> 16);
  return __builtin_bit_cast(__bf16, h);
}

// Stage `chunks` 16-byte chunks from global to LDS with all 256 threads via
// the gfx1250 async-DMA path (ASYNCcnt): no VGPR return slots occupied.
static __device__ __forceinline__ void stage_panel(const __bf16* g, __bf16* l,
                                                   int chunks, int tid) {
  const uint4* src = (const uint4*)g;
  uint4* dst = (uint4*)l;
#if HAVE_ASYNC_LDS
  for (int i = tid; i < chunks; i += 256)
    __builtin_amdgcn_global_load_async_to_lds_b128(
        (v4i*)(src + i),
        (__attribute__((address_space(3))) v4i*)(dst + i), 0, 0);
  __builtin_amdgcn_s_wait_asynccnt(0);
#else
  for (int i = tid; i < chunks; i += 256) dst[i] = src[i];
#endif
}

// A-fragment (16-bit 16x32): lanes 0-15 -> K=0..15,
// lanes 16-31 -> K=16..31, row M = lane%16 (matches B-fragment order).
static __device__ __forceinline__ v16bf load_a_frag(const __bf16* ldsA, int K,
                                                    int k0, int lane) {
  const int m    = lane & 15;
  const int base = (lane >> 4) << 4;        // 0 or 16
  const __bf16* p = ldsA + m * K + k0 + base;
  v8bf lo = *(const v8bf*)p;
  v8bf hi = *(const v8bf*)(p + 8);
  return __builtin_shufflevector(lo, hi, 0,1,2,3,4,5,6,7,8,9,10,11,12,13,14,15);
}

// B-fragment (32x16): column n = W row n (W stored [N,K] row-major), so a
// lane's 16 K-values are one contiguous 32-byte load.
static __device__ __forceinline__ v16bf load_b_frag(const __bf16* B, int K,
                                                    int n0, int k0, int lane) {
  const int n  = n0 + (lane & 15);
  const int kb = k0 + ((lane >> 4) << 4);   // 0 or 16
  return *(const v16bf*)(B + (size_t)n * K + kb);
}

// ---------------------------------------------------------------------------
// C[M,N] = A[M,K] @ B[N,K]^T + bias   (A,B bf16; C f32).  Block = 8 waves,
// each wave owns one 16x16 tile of a 16x128 block tile.  A K-panel staged
// once in LDS via async DMA, B streamed from global (L2-resident weights).
// K is a compile-time constant; software pipeline with peeled epilogue so the
// steady state is branch-free: issue frags(k+1), partial-wait, WMMA(k).
// ---------------------------------------------------------------------------
template <int K>
__global__ __launch_bounds__(256) void wmma_gemm_atb(
    const __bf16* __restrict__ A, const __bf16* __restrict__ B,
    const float* __restrict__ bias, float* __restrict__ C, int N) {
  __shared__ __bf16 ldsA[16 * K];
  const int tid = threadIdx.x;
  const int m0  = blockIdx.y * 16;
  const int n0  = blockIdx.x * 128 + (tid >> 5) * 16;

  stage_panel(A + (size_t)m0 * K, ldsA, (16 * K) >> 3, tid);
  __syncthreads();

  const int lane = tid & 31;
  v8f acc = {};
  v16bf a = load_a_frag(ldsA, K, 0, lane);
  v16bf b = load_b_frag(B, K, n0, 0, lane);
  for (int k0 = 32; k0 < K; k0 += 32) {
    v16bf an = load_a_frag(ldsA, K, k0, lane);
    v16bf bn = load_b_frag(B, K, n0, k0, lane);
    if (k0 + 32 < K)   // static per unrolled iteration (K is constexpr)
      __builtin_prefetch(B + (size_t)(n0 + (lane & 15)) * K + k0 + 32, 0, 1);
    acc = __builtin_amdgcn_wmma_f32_16x16x32_bf16(false, a, false, b,
                                                  (short)0, acc, false, false);
    a = an; b = bn;
  }
  acc = __builtin_amdgcn_wmma_f32_16x16x32_bf16(false, a, false, b,
                                                (short)0, acc, false, false);

  // C/D layout: VGPR r -> M = r + 8*(lane>=16), N = n0 + lane%16
  const int n  = n0 + (lane & 15);
  const int mb = (lane >> 4) << 3;
  const float bn = bias ? bias[n] : 0.0f;
  float* Cp = C + (size_t)(m0 + mb) * N + n;
#pragma unroll
  for (int r = 0; r < 8; ++r) Cp[(size_t)r * N] = acc[r] + bn;
}

// ---------------------------------------------------------------------------
// Persistent LSTM recurrence.  Grid = 32 blocks; block b owns hidden units
// [32b,32b+32).  Per step: h (bf16, double-buffered) -> LDS via async DMA,
// each of 8 waves computes one 16x16 gate tile of h @ W_hh^T via WMMA, adds
// precomputed input-projection G, gates fuse through LDS, then a device-scope
// barrier.  One barrier per step thanks to the h ping-pong.
// ---------------------------------------------------------------------------
__global__ __launch_bounds__(256) void lstm_recurrence(
    const __bf16* __restrict__ Whh,    // [NGATE, HID] bf16
    const float*  __restrict__ G,      // [NTOK, NGATE] = x@W_ih^T + b_ih + b_hh
    __bf16* __restrict__ hping,        // [2][BATCH*HID] bf16, buf0 initialized
    float*  __restrict__ c,            // [BATCH*HID] f32
    __bf16* __restrict__ Hseq,         // [NTOK, HID] bf16 layer output
    float* __restrict__ hT, float* __restrict__ cT,  // [BATCH*HID] finals
    unsigned* __restrict__ bar) {      // {count, generation}, zeroed
  __shared__ __bf16 ldsA[BATCH * HID];     // 32KB
  __shared__ float  ldsGate[BATCH * 128];  // 8KB
  const int tid  = threadIdx.x;
  const int lane = tid & 31;
  const int w    = tid >> 5;               // wave 0..7
  const int u0   = blockIdx.x * 32;        // this block's hidden-unit slice
  // wave -> gate tile: gate g = w>>1, sub-slice (w&1)*16
  const int n0 = (w >> 1) * HID + u0 + (w & 1) * 16;

  for (int t = 0; t < SEQ; ++t) {
    const __bf16* hin  = hping + (size_t)(t & 1) * (BATCH * HID);
    __bf16*       hout = hping + (size_t)((t + 1) & 1) * (BATCH * HID);

    stage_panel(hin, ldsA, (BATCH * HID) >> 3, tid);
    __syncthreads();

    v8f acc = {};
    v16bf a = load_a_frag(ldsA, HID, 0, lane);
    v16bf b = load_b_frag(Whh, HID, n0, 0, lane);
    for (int k0 = 32; k0 < HID; k0 += 32) {
      v16bf an = load_a_frag(ldsA, HID, k0, lane);
      v16bf bn = load_b_frag(Whh, HID, n0, k0, lane);
      acc = __builtin_amdgcn_wmma_f32_16x16x32_bf16(false, a, false, b,
                                                    (short)0, acc, false, false);
      a = an; b = bn;
    }
    acc = __builtin_amdgcn_wmma_f32_16x16x32_bf16(false, a, false, b,
                                                  (short)0, acc, false, false);

    {  // add precomputed input gates, park pre-activations in LDS
      const int n  = n0 + (lane & 15);
      const int mb = (lane >> 4) << 3;
      const int lc = w * 16 + (lane & 15);   // == g*32 + u_local
#pragma unroll
      for (int r = 0; r < 8; ++r) {
        const int m = mb + r;
        ldsGate[m * 128 + lc] = acc[r] + G[(size_t)(m * SEQ + t) * NGATE + n];
      }
    }
    __syncthreads();

    // elementwise cell update: 16 batch x 32 units -> 2 per thread
    for (int e = tid; e < BATCH * 32; e += 256) {
      const int m  = e >> 5;
      const int ul = e & 31;
      const int u  = u0 + ul;
      float gi = ldsGate[m * 128 +   0 + ul];
      float gf = ldsGate[m * 128 +  32 + ul];
      float gg = ldsGate[m * 128 +  64 + ul];
      float go = ldsGate[m * 128 +  96 + ul];
      gi = 1.0f / (1.0f + __expf(-gi));
      gf = 1.0f / (1.0f + __expf(-gf));
      gg = tanhf(gg);
      go = 1.0f / (1.0f + __expf(-go));
      const float cn = gf * c[m * HID + u] + gi * gg;
      const float hn = go * tanhf(cn);
      c[m * HID + u] = cn;
      const __bf16 hb = f2bf(hn);
      hout[m * HID + u] = hb;
      Hseq[(size_t)(m * SEQ + t) * HID + u] = hb;
      if (t == SEQ - 1) { hT[m * HID + u] = hn; cT[m * HID + u] = cn; }
    }

    // device-scope sense-reversing barrier (32 co-resident blocks)
    __syncthreads();
    if (tid == 0) {
      __threadfence();
      const unsigned g = __hip_atomic_load(bar + 1, __ATOMIC_RELAXED,
                                           __HIP_MEMORY_SCOPE_AGENT);
      const unsigned old = __hip_atomic_fetch_add(bar, 1u, __ATOMIC_ACQ_REL,
                                                  __HIP_MEMORY_SCOPE_AGENT);
      if (old + 1u == gridDim.x) {
        __hip_atomic_store(bar, 0u, __ATOMIC_RELAXED, __HIP_MEMORY_SCOPE_AGENT);
        __hip_atomic_store(bar + 1, g + 1u, __ATOMIC_RELEASE,
                           __HIP_MEMORY_SCOPE_AGENT);
      } else {
        while (__hip_atomic_load(bar + 1, __ATOMIC_ACQUIRE,
                                 __HIP_MEMORY_SCOPE_AGENT) == g)
          __builtin_amdgcn_s_sleep(2);
      }
    }
    __syncthreads();
  }
}

// ---------------------------- small helper kernels --------------------------

__global__ void f32_to_bf16_k(const float* __restrict__ s,
                              __bf16* __restrict__ d, int n) {
  const int i = blockIdx.x * blockDim.x + threadIdx.x;
  if (i < n) d[i] = f2bf(s[i]);
}

__global__ void bias_sum_k(const float* __restrict__ a,
                           const float* __restrict__ b,
                           float* __restrict__ o, int n) {
  const int i = blockIdx.x * blockDim.x + threadIdx.x;
  if (i < n) o[i] = a[i] + b[i];
}

__global__ void embed_k(const int* __restrict__ x, const float* __restrict__ emb,
                        __bf16* __restrict__ X) {
  const int i = blockIdx.x * blockDim.x + threadIdx.x;   // over NTOK*EMB
  if (i < NTOK * EMB) {
    const int tok = i >> 9;          // /EMB
    const int e   = i & (EMB - 1);
    X[i] = f2bf(emb[(size_t)x[tok] * EMB + e]);
  }
}

__global__ void init_state_k(const float* __restrict__ h0,
                             const float* __restrict__ c0,
                             __bf16* __restrict__ hping, float* __restrict__ c,
                             unsigned* __restrict__ bar) {
  const int i = blockIdx.x * blockDim.x + threadIdx.x;
  if (i < BATCH * HID) {
    hping[i] = f2bf(h0[i]);          // buffer 0 is read at t=0
    c[i]     = c0[i];
  }
  if (i < 2) bar[i] = 0u;
}

// ---------------------------------------------------------------------------

extern "C" void kernel_launch(void* const* d_in, const int* in_sizes, int n_in,
                              void* d_out, int out_size, void* d_ws,
                              size_t ws_size, hipStream_t stream) {
  (void)in_sizes; (void)n_in; (void)out_size; (void)ws_size;
  const int*   x    = (const int*)  d_in[0];
  const float* h0   = (const float*)d_in[1];
  const float* c0   = (const float*)d_in[2];
  const float* emb  = (const float*)d_in[3];
  const float* Wih0 = (const float*)d_in[4];
  const float* Whh0 = (const float*)d_in[5];
  const float* bih0 = (const float*)d_in[6];
  const float* bhh0 = (const float*)d_in[7];
  const float* Wih1 = (const float*)d_in[8];
  const float* Whh1 = (const float*)d_in[9];
  const float* bih1 = (const float*)d_in[10];
  const float* bhh1 = (const float*)d_in[11];
  const float* fcW  = (const float*)d_in[12];
  const float* fcb  = (const float*)d_in[13];
  float* out = (float*)d_out;

  // workspace carve (256B aligned)
  char* p = (char*)d_ws;
  auto alloc = [&](size_t bytes) {
    char* r = p; p += (bytes + 255) & ~(size_t)255; return r;
  };
  __bf16* Wih0b = (__bf16*)alloc((size_t)NGATE * EMB * 2);
  __bf16* Whh0b = (__bf16*)alloc((size_t)NGATE * HID * 2);
  __bf16* Wih1b = (__bf16*)alloc((size_t)NGATE * HID * 2);
  __bf16* Whh1b = (__bf16*)alloc((size_t)NGATE * HID * 2);
  __bf16* fcWb  = (__bf16*)alloc((size_t)VOCAB * HID * 2);
  __bf16* Xb    = (__bf16*)alloc((size_t)NTOK * EMB * 2);
  float*  Gbuf  = (float*) alloc((size_t)NTOK * NGATE * 4);  // reused L0/L1
  __bf16* Hseq0 = (__bf16*)alloc((size_t)NTOK * HID * 2);
  __bf16* Hseq1 = (__bf16*)alloc((size_t)NTOK * HID * 2);
  __bf16* hp0   = (__bf16*)alloc((size_t)2 * BATCH * HID * 2);
  float*  cc0   = (float*) alloc((size_t)BATCH * HID * 4);
  __bf16* hp1   = (__bf16*)alloc((size_t)2 * BATCH * HID * 2);
  float*  cc1   = (float*) alloc((size_t)BATCH * HID * 4);
  float*  bs0   = (float*) alloc((size_t)NGATE * 4);
  float*  bs1   = (float*) alloc((size_t)NGATE * 4);
  unsigned* bar0 = (unsigned*)alloc(256);
  unsigned* bar1 = (unsigned*)alloc(256);

  const size_t LOGITS = (size_t)NTOK * VOCAB;   // 131,072,000
  float* hT0 = out + LOGITS;
  float* hT1 = out + LOGITS + BATCH * HID;
  float* cT0 = out + LOGITS + 2 * BATCH * HID;
  float* cT1 = out + LOGITS + 3 * BATCH * HID;

  // weight / bias conversion (one pass; bf16 weights then live in L2)
  f32_to_bf16_k<<<(NGATE * EMB + 255) / 256, 256, 0, stream>>>(Wih0, Wih0b, NGATE * EMB);
  f32_to_bf16_k<<<(NGATE * HID + 255) / 256, 256, 0, stream>>>(Whh0, Whh0b, NGATE * HID);
  f32_to_bf16_k<<<(NGATE * HID + 255) / 256, 256, 0, stream>>>(Wih1, Wih1b, NGATE * HID);
  f32_to_bf16_k<<<(NGATE * HID + 255) / 256, 256, 0, stream>>>(Whh1, Whh1b, NGATE * HID);
  f32_to_bf16_k<<<(VOCAB * HID + 255) / 256, 256, 0, stream>>>(fcW, fcWb, VOCAB * HID);
  bias_sum_k<<<(NGATE + 255) / 256, 256, 0, stream>>>(bih0, bhh0, bs0, NGATE);
  bias_sum_k<<<(NGATE + 255) / 256, 256, 0, stream>>>(bih1, bhh1, bs1, NGATE);

  embed_k<<<(NTOK * EMB + 255) / 256, 256, 0, stream>>>(x, emb, Xb);
  init_state_k<<<(BATCH * HID + 255) / 256, 256, 0, stream>>>(
      h0 + 0 * BATCH * HID, c0 + 0 * BATCH * HID, hp0, cc0, bar0);
  init_state_k<<<(BATCH * HID + 255) / 256, 256, 0, stream>>>(
      h0 + 1 * BATCH * HID, c0 + 1 * BATCH * HID, hp1, cc1, bar1);

  // Layer 0: input projection (all timesteps at once), then recurrence
  wmma_gemm_atb<EMB><<<dim3(NGATE / 128, NTOK / 16), 256, 0, stream>>>(
      Xb, Wih0b, bs0, Gbuf, NGATE);
  lstm_recurrence<<<32, 256, 0, stream>>>(Whh0b, Gbuf, hp0, cc0, Hseq0,
                                          hT0, cT0, bar0);

  // Layer 1
  wmma_gemm_atb<HID><<<dim3(NGATE / 128, NTOK / 16), 256, 0, stream>>>(
      Hseq0, Wih1b, bs1, Gbuf, NGATE);
  lstm_recurrence<<<32, 256, 0, stream>>>(Whh1b, Gbuf, hp1, cc1, Hseq1,
                                          hT1, cT1, bar1);

  // Final FC straight into d_out (dominant cost: 524 MB logits write)
  wmma_gemm_atb<HID><<<dim3(VOCAB / 128, NTOK / 16), 256, 0, stream>>>(
      Hseq1, fcWb, fcb, out, VOCAB);
}
